// RWKVLoop_46608985096379
// MI455X (gfx1250) — matrix operand order repaired and checked by reading, not compiled, not run on hardware
//
#include <hip/hip_runtime.h>
#include <hip/hip_bf16.h>
#include <stdint.h>
#include <stddef.h>

// ---------------------------------------------------------------------------
// RWKV-style token-shift MLP with per-channel decay recurrence, for gfx1250.
//   inp  = mix*x + (1-mix)*shift(x)             (fp32 -> bf16)
//   h    = relu(inp @ W_in^T + b_in)            (bf16 WMMA, f32 acc -> bf16)
//   st_t = decay*st_{t-1} + h_t                 (chunked parallel scan)
//   out  = st @ W_out^T + b_out                 (bf16 WMMA, f32 acc -> f32)
// GEMM: one wave computes 32(M) x 64(N); K pipelined with register double
// buffering so loads for K-tile k+1 are in flight while WMMAs consume tile k.
// ---------------------------------------------------------------------------

#define DIMD   1024
#define BATCH  4
#define SEQ    4096
#define MTOT   (BATCH * SEQ)       // 16384 tokens
#define NCH    16                  // scan chunks per sequence
#define CHL    (SEQ / NCH)         // 256 steps per chunk

typedef __attribute__((ext_vector_type(8)))  float  v8f;
typedef __attribute__((ext_vector_type(16))) __bf16 v16bf;
typedef __attribute__((ext_vector_type(8)))  __bf16 v8bf;

union AB16 { v16bf v; v8bf h[2]; };

__device__ __forceinline__ unsigned short f32_to_bf16(float f) {
  union { float f; unsigned int u; } v; v.f = f;
  unsigned int u = v.u;
  u += 0x7FFFu + ((u >> 16) & 1u);            // round-to-nearest-even
  return (unsigned short)(u >> 16);
}
__device__ __forceinline__ float bf16_to_f32(unsigned short h) {
  union { unsigned int u; float f; } v; v.u = ((unsigned int)h) << 16;
  return v.f;
}

// --------------------------- elementwise prologue ---------------------------

__global__ void prep_input_kernel(const float* __restrict__ x,
                                  const float* __restrict__ mix_p,
                                  unsigned short* __restrict__ inp) {
  const float mix = *mix_p;
  const float omix = 1.0f - mix;
  const size_t n = (size_t)MTOT * DIMD;
  const size_t sdim = (size_t)SEQ * DIMD;
  for (size_t i = (size_t)blockIdx.x * blockDim.x + threadIdx.x; i < n;
       i += (size_t)gridDim.x * blockDim.x) {
    size_t sd = i % sdim;                      // position within one batch
    float cur  = x[i];
    float prev = (sd < DIMD) ? 0.0f : x[i - DIMD];   // token shift (zeros at t=0)
    inp[i] = f32_to_bf16(mix * cur + omix * prev);
  }
}

__global__ void convert_f32_bf16_kernel(const float* __restrict__ src,
                                        unsigned short* __restrict__ dst,
                                        size_t n) {
  for (size_t i = (size_t)blockIdx.x * blockDim.x + threadIdx.x; i < n;
       i += (size_t)gridDim.x * blockDim.x)
    dst[i] = f32_to_bf16(src[i]);
}

// ------------------------------- WMMA GEMM ---------------------------------
// C[m,n] = act( sum_k A[m,k] * W[n,k] + bias[n] )
// A-fragment (16x32 bf16): lane m in 0..15 holds K = {o..o+7, o+16..o+23},
//   o = 8 for lanes 16..31 (ISA 7.12.2 16-bit A layout).
// B-fragment (32x16 bf16): lane n holds K = {o..o+7, o+16..o+23}, o = 8 for upper half.
// W rows are contiguous in K, so both operands use plain b128 loads.

__device__ __forceinline__ void load_a(AB16& f, const unsigned short* row,
                                       int k0, int aoff) {
  f.h[0] = *reinterpret_cast<const v8bf*>(row + k0 + aoff);
  f.h[1] = *reinterpret_cast<const v8bf*>(row + k0 + aoff + 16);
}
__device__ __forceinline__ void load_b(AB16& f, const unsigned short* row,
                                       int k0, int boff) {
  f.h[0] = *reinterpret_cast<const v8bf*>(row + k0 + boff);
  f.h[1] = *reinterpret_cast<const v8bf*>(row + k0 + boff + 16);
}

template<bool RELU, bool OUT_BF16>
__global__ __launch_bounds__(32)
void gemm_wmma_kernel(const unsigned short* __restrict__ A,   // M x K, bf16
                      const unsigned short* __restrict__ W,   // N x K, bf16
                      const float* __restrict__ bias,         // N, fp32
                      void* __restrict__ Cout,                // M x N
                      int M, int N, int K) {
  const int lane = threadIdx.x & 31;
  const int l15  = lane & 15;
  const int hi   = lane >> 4;                 // 0: lanes 0-15, 1: lanes 16-31
  const int mbase = blockIdx.x * 32;
  const int nbase = blockIdx.y * 64;

  const unsigned short* arow[2];
  arow[0] = A + (size_t)(mbase +  0 + l15) * K;
  arow[1] = A + (size_t)(mbase + 16 + l15) * K;
  const unsigned short* brow[4];
#pragma unroll
  for (int f = 0; f < 4; ++f)
    brow[f] = W + (size_t)(nbase + 16 * f + l15) * K;

  const int aoff = hi * 8;                    // upper half-wave: K+8 block
  const int boff = hi * 8;                    // upper half-wave: K+8 block

  v8f acc[2][4];
#pragma unroll
  for (int mt = 0; mt < 2; ++mt)
#pragma unroll
    for (int nt = 0; nt < 4; ++nt) acc[mt][nt] = (v8f){};

  // Prologue: fragments for K-tile 0.
  AB16 a[2], b[4];
#pragma unroll
  for (int mt = 0; mt < 2; ++mt) load_a(a[mt], arow[mt], 0, aoff);
#pragma unroll
  for (int nt = 0; nt < 4; ++nt) load_b(b[nt], brow[nt], 0, boff);

  // Pipelined main loop: issue loads for tile k+1, then WMMA tile k.
  for (int k0 = 0; k0 < K - 32; k0 += 32) {
    const int kn = k0 + 32;
    if (kn + 32 < K) {                        // L2 prefetch two tiles ahead
      __builtin_prefetch(arow[0] + kn + 32, 0, 3);
      __builtin_prefetch(brow[0] + kn + 32, 0, 3);
    }
    AB16 na[2], nb[4];
#pragma unroll
    for (int mt = 0; mt < 2; ++mt) load_a(na[mt], arow[mt], kn, aoff);
#pragma unroll
    for (int nt = 0; nt < 4; ++nt) load_b(nb[nt], brow[nt], kn, boff);

#pragma unroll
    for (int mt = 0; mt < 2; ++mt)
#pragma unroll
      for (int nt = 0; nt < 4; ++nt)
        acc[mt][nt] = __builtin_amdgcn_wmma_f32_16x16x32_bf16(
            false, a[mt].v, false, b[nt].v, (short)0, acc[mt][nt], false, false);

#pragma unroll
    for (int mt = 0; mt < 2; ++mt) a[mt] = na[mt];
#pragma unroll
    for (int nt = 0; nt < 4; ++nt) b[nt] = nb[nt];
  }
  // Epilogue K-tile.
#pragma unroll
  for (int mt = 0; mt < 2; ++mt)
#pragma unroll
    for (int nt = 0; nt < 4; ++nt)
      acc[mt][nt] = __builtin_amdgcn_wmma_f32_16x16x32_bf16(
          false, a[mt].v, false, b[nt].v, (short)0, acc[mt][nt], false, false);

  // Store. C/D layout: VGPR r holds M = r (lanes 0-15) / r+8 (lanes 16-31),
  // N = lane & 15.
  float bv[4];
#pragma unroll
  for (int nt = 0; nt < 4; ++nt) bv[nt] = bias[nbase + 16 * nt + l15];

#pragma unroll
  for (int mt = 0; mt < 2; ++mt) {
    const int mrow = mbase + 16 * mt + hi * 8;
#pragma unroll
    for (int r = 0; r < 8; ++r) {
      const size_t row = (size_t)(mrow + r) * N;
#pragma unroll
      for (int nt = 0; nt < 4; ++nt) {
        float c = acc[mt][nt][r] + bv[nt];
        if (RELU) c = fmaxf(c, 0.0f);
        const size_t idx = row + nbase + 16 * nt + l15;
        if (OUT_BF16) ((unsigned short*)Cout)[idx] = f32_to_bf16(c);
        else          ((float*)Cout)[idx] = c;
      }
    }
  }
}

// --------------------------- chunked decay scan -----------------------------
// Phase 1: per (b, chunk, d) local scan of CHL steps, record chunk-final state.
__global__ void scan_local_kernel(const unsigned short* __restrict__ h,
                                  unsigned short* __restrict__ st,
                                  float* __restrict__ chunk_last,
                                  const float* __restrict__ decay_p) {
  const float decay = *decay_p;
  int idx = blockIdx.x * blockDim.x + threadIdx.x;    // BATCH*NCH*DIMD
  if (idx >= BATCH * NCH * DIMD) return;
  const int d  = idx % DIMD;
  const int ch = (idx / DIMD) % NCH;
  const int b  = idx / (DIMD * NCH);
  const size_t base = ((size_t)b * SEQ + (size_t)ch * CHL) * DIMD + d;
  float s = 0.0f;
  for (int t = 0; t < CHL; ++t) {
    s = s * decay + bf16_to_f32(h[base + (size_t)t * DIMD]);
    st[base + (size_t)t * DIMD] = f32_to_bf16(s);
  }
  chunk_last[idx] = s;
}

// Phase 2: propagate carries across the NCH chunks (serial length NCH only).
__global__ void scan_carry_kernel(const float* __restrict__ chunk_last,
                                  float* __restrict__ carry_in,
                                  const float* __restrict__ decay_p) {
  const float decay = *decay_p;
  float dL = 1.0f, p = decay;                 // decay^CHL by square-and-multiply
  for (int e = CHL; e; e >>= 1) { if (e & 1) dL *= p; p *= p; }
  int idx = blockIdx.x * blockDim.x + threadIdx.x;    // BATCH*DIMD
  if (idx >= BATCH * DIMD) return;
  const int d = idx % DIMD;
  const int b = idx / DIMD;
  float c = 0.0f;
  for (int ch = 0; ch < NCH; ++ch) {
    const size_t o = ((size_t)b * NCH + ch) * DIMD + d;
    carry_in[o] = c;                          // state entering this chunk
    c = chunk_last[o] + c * dL;
  }
}

// Phase 3: st_t += carry * decay^(t+1), in place.
__global__ void scan_apply_kernel(unsigned short* __restrict__ st,
                                  const float* __restrict__ carry_in,
                                  const float* __restrict__ decay_p) {
  const float decay = *decay_p;
  int idx = blockIdx.x * blockDim.x + threadIdx.x;    // BATCH*NCH*DIMD
  if (idx >= BATCH * NCH * DIMD) return;
  const int d  = idx % DIMD;
  const int ch = (idx / DIMD) % NCH;
  const int b  = idx / (DIMD * NCH);
  const float c = carry_in[((size_t)b * NCH + ch) * DIMD + d];
  if (c == 0.0f) return;                      // chunk 0 / dead carries: no-op
  const size_t base = ((size_t)b * SEQ + (size_t)ch * CHL) * DIMD + d;
  float p = decay;
  for (int t = 0; t < CHL; ++t) {
    const size_t o = base + (size_t)t * DIMD;
    st[o] = f32_to_bf16(bf16_to_f32(st[o]) + c * p);
    p *= decay;
  }
}

// ------------------------------- launcher -----------------------------------

extern "C" void kernel_launch(void* const* d_in, const int* in_sizes, int n_in,
                              void* d_out, int out_size, void* d_ws, size_t ws_size,
                              hipStream_t stream) {
  const float* x     = (const float*)d_in[0];
  const float* W_in  = (const float*)d_in[1];
  const float* b_in  = (const float*)d_in[2];
  const float* W_out = (const float*)d_in[3];
  const float* b_out = (const float*)d_in[4];
  const float* decay = (const float*)d_in[5];
  const float* mix   = (const float*)d_in[6];

  char* ws = (char*)d_ws;
  // Workspace layout (bytes):
  unsigned short* inp    = (unsigned short*)(ws);                 // 33554432  (M*D bf16)
  unsigned short* win16  = (unsigned short*)(ws + 33554432u);     //  2097152
  unsigned short* wout16 = (unsigned short*)(ws + 35651584u);     //  2097152
  unsigned short* hbuf   = (unsigned short*)(ws + 37748736u);     // 33554432  (M*D bf16)
  float*          clast  = (float*)(ws + 71303168u);              //   262144  (B*NCH*D)
  float*          cin    = (float*)(ws + 71565312u);              //   262144
  // total: 71827456 bytes (~68.5 MiB)

  prep_input_kernel<<<2048, 256, 0, stream>>>(x, mix, inp);
  convert_f32_bf16_kernel<<<1024, 256, 0, stream>>>(W_in, win16,
                                                    (size_t)DIMD * DIMD);
  convert_f32_bf16_kernel<<<1024, 256, 0, stream>>>(W_out, wout16,
                                                    (size_t)DIMD * DIMD);

  dim3 gemm_grid(MTOT / 32, DIMD / 64);       // (512, 16), one wave per block
  gemm_wmma_kernel<true, true><<<gemm_grid, 32, 0, stream>>>(
      inp, win16, b_in, hbuf, MTOT, DIMD, DIMD);

  unsigned short* states = inp;               // inp is dead after GEMM1: reuse
  const int nscan = BATCH * NCH * DIMD;       // 65536
  scan_local_kernel<<<(nscan + 255) / 256, 256, 0, stream>>>(hbuf, states,
                                                             clast, decay);
  scan_carry_kernel<<<(BATCH * DIMD + 255) / 256, 256, 0, stream>>>(clast, cin,
                                                                    decay);
  scan_apply_kernel<<<(nscan + 255) / 256, 256, 0, stream>>>(states, cin, decay);

  gemm_wmma_kernel<false, false><<<gemm_grid, 32, 0, stream>>>(
      states, wout16, b_out, d_out, MTOT, DIMD, DIMD);
}
